// MultiHeadAttention_15573551415608
// MI455X (gfx1250) — matrix operand order repaired and checked by reading, not compiled, not run on hardware
//
#include <hip/hip_runtime.h>
#include <hip/hip_bf16.h>

typedef __attribute__((ext_vector_type(16))) _Float16 v16h;
typedef __attribute__((ext_vector_type(8)))  _Float16 h8;
typedef __attribute__((ext_vector_type(8)))  float    v8f;
typedef __attribute__((ext_vector_type(4)))  float    v4f;

#define NBATCH 4
#define SEQ    2048
#define DIM    256
#define NH     8
#define DH     32

// ---- fragment load helpers -------------------------------------------------
static __device__ __forceinline__ void ld8_h(const _Float16* p, v16h& d, int base) {
  h8 t = *(const h8*)p;
#pragma unroll
  for (int i = 0; i < 8; ++i) d[base + i] = t[i];
}
static __device__ __forceinline__ void ld8_f2h(const float* p, v16h& d, int base) {
  v4f a = *(const v4f*)p;
  v4f b = *(const v4f*)(p + 4);
#pragma unroll
  for (int i = 0; i < 4; ++i) { d[base + i] = (_Float16)a[i]; d[base + 4 + i] = (_Float16)b[i]; }
}
static __device__ __forceinline__ v8f wmma_f16(v16h a, v16h b, v8f c) {
  return __builtin_amdgcn_wmma_f32_16x16x32_f16(false, a, false, b, (short)0, c, false, false);
}

// ---- phase 1: Q/K/V projections, f32 -> f16, WMMA GEMM ---------------------
// tile ids: 512 mtiles x 16 ntiles x 3 weights = 24576 wave-tiles, 8 waves/WG
__global__ __launch_bounds__(256) void qkv_kernel(
    const float* __restrict__ X,
    const float* __restrict__ Wq, const float* __restrict__ Wk, const float* __restrict__ Wv,
    _Float16* __restrict__ qb, _Float16* __restrict__ kb, _Float16* __restrict__ vtb) {
  const int lane = threadIdx.x & 31;
  const int wid  = threadIdx.x >> 5;
  const int tid  = blockIdx.x * 8 + wid;
  const int ntile = tid & 15;
  const int mtile = (tid >> 4) & 511;
  const int w     = tid >> 13;                 // 0=Q 1=K 2=V
  const float* W = (w == 0) ? Wq : (w == 1) ? Wk : Wv;
  const int l15  = lane & 15;
  const int hf   = lane >> 4;
  const int kh8  = hf << 3;                    // A-fragment k sub-block
  const int kh16 = hf << 4;                    // unused; B also uses kh8
  const int m0 = mtile << 4, n0 = ntile << 4;
  const float* xrow = X + (size_t)(m0 + l15) * DIM;
  const float* wrow = W + (size_t)(n0 + l15) * DIM;
  v8f acc = {};
#pragma unroll
  for (int kt = 0; kt < DIM; kt += 32) {
    v16h a, b;
    ld8_f2h(xrow + kt + kh8,      a, 0);
    ld8_f2h(xrow + kt + 16 + kh8, a, 8);
    ld8_f2h(wrow + kt + kh8,      b, 0);
    ld8_f2h(wrow + kt + 16 + kh8, b, 8);
    acc = wmma_f16(a, b, acc);
  }
  const int ncol = n0 + l15;
  const int h = ncol >> 5, dcol = ncol & 31;
#pragma unroll
  for (int r = 0; r < 8; ++r) {
    int m = m0 + r + (hf << 3);
    int bb = m >> 11, s = m & (SEQ - 1);
    _Float16 val = (_Float16)acc[r];
    if (w == 2) {
      // V stored transposed: [B,H,dh,S]
      vtb[(((size_t)bb * NH + h) * DH + dcol) * SEQ + s] = val;
    } else {
      _Float16* dst = (w == 0) ? qb : kb;      // [B,H,S,dh]
      dst[(((size_t)bb * NH + h) * SEQ + s) * DH + dcol] = val;
    }
  }
}

// ---- phase 2: flash attention, transposed score / PV GEMMs -----------------
// wave = (b,h,qtile16): 4*8*128 = 4096 waves = 512 WGs x 8 waves
__global__ __launch_bounds__(256) void attn_kernel(
    const _Float16* __restrict__ qb, const _Float16* __restrict__ kb,
    const _Float16* __restrict__ vtb, _Float16* __restrict__ ao) {
  const int lane = threadIdx.x & 31;
  const int wid  = threadIdx.x >> 5;
  const int id   = blockIdx.x * 8 + wid;
  const int qtile = id & 127;
  const int h = (id >> 7) & 7;
  const int b = id >> 10;
  const int l15  = lane & 15;
  const int hf   = lane >> 4;
  const int kh8  = hf << 3;
  const int kh16 = hf << 4;
  const _Float16* Qp = qb  + (((size_t)b * NH + h) * SEQ + (qtile << 4)) * DH;
  const _Float16* Kp = kb  + ((size_t)b * NH + h) * SEQ * DH;
  const _Float16* Vp = vtb + ((size_t)b * NH + h) * DH * SEQ;

  // B fragment: Q^T (d x q): lane -> q = l15, elements -> d = kh8 + e (e<8), 16 + kh8 + (e-8) (e>=8)
  v16h bq;
  ld8_h(Qp + l15 * DH + kh8,      bq, 0);
  ld8_h(Qp + l15 * DH + 16 + kh8, bq, 8);

  v8f o0 = {}, o1 = {};                        // O^T tiles: d 0-15 / 16-31
  float mrun = -3.0e38f, lrun = 0.0f;
  // softmax in base-2 domain: cscale = (1/sqrt(S)) * log2(e); softmax is
  // invariant to the base change since we normalize by the running sum.
  const float cscale = 0.022097086912079608f * 1.4426950408889634f;

  for (int kt = 0; kt < SEQ; kt += 32) {
    // prefetch K/V two chunks ahead (uniform branch; EXEC untouched)
    if (kt + 64 < SEQ) {
      __builtin_prefetch(Kp + (size_t)(kt + 64 + lane) * DH, 0, 1);
      __builtin_prefetch(Vp + (size_t)lane * SEQ + kt + 64, 0, 1);
    }
    // scores^T tiles: A = K-tile (kk x d), B = Q^T -> D lane: q=l15, m=kk
    v16h ak0, ak1;
    ld8_h(Kp + (kt + l15) * DH + kh8,           ak0, 0);
    ld8_h(Kp + (kt + l15) * DH + 16 + kh8,      ak0, 8);
    ld8_h(Kp + (kt + 16 + l15) * DH + kh8,      ak1, 0);
    ld8_h(Kp + (kt + 16 + l15) * DH + 16 + kh8, ak1, 8);
    v8f z = {};
    v8f s0 = wmma_f16(ak0, bq, z);
    v8f s1 = wmma_f16(ak1, bq, z);

    // online softmax; each lane owns one query column (partner = lane^16)
    float cmax = -3.0e38f;
#pragma unroll
    for (int r = 0; r < 8; ++r) cmax = fmaxf(cmax, fmaxf(s0[r], s1[r]));
    cmax *= cscale;
    cmax = fmaxf(cmax, __shfl_xor(cmax, 16, 32));
    float mnew = fmaxf(mrun, cmax);
    float alpha = exp2f(mrun - mnew);

    // p = 2^(s*cscale - mnew); pack (p_lowblock, p_highblock) as f16 pair and
    // exchange with partner lane in a single 32-bit shuffle per register.
    v16h bp;
    float csum = 0.0f;
    const bool low = (hf == 0);
#pragma unroll
    for (int r = 0; r < 8; ++r) {
      float p0 = exp2f(fmaf(s0[r], cscale, -mnew));
      float p1 = exp2f(fmaf(s1[r], cscale, -mnew));
      csum += p0 + p1;
      union { unsigned u; _Float16 hh[2]; } own, oth;
      own.hh[0] = (_Float16)p0;
      own.hh[1] = (_Float16)p1;
      oth.u = (unsigned)__shfl_xor((int)own.u, 16, 32);
      bp[r]     = low ? own.hh[0] : oth.hh[1];
      bp[8 + r] = low ? oth.hh[0] : own.hh[1];
    }
    csum += __shfl_xor(csum, 16, 32);
    lrun = lrun * alpha + csum;
    mrun = mnew;
#pragma unroll
    for (int r = 0; r < 8; ++r) { o0[r] *= alpha; o1[r] *= alpha; }

    // O^T += V^T-tile * P^T  (A = V^T rows d, K=32 keys)
    v16h av0, av1;
    ld8_h(Vp + (size_t)l15 * SEQ + kt + kh16,            av0, 0);
    ld8_h(Vp + (size_t)l15 * SEQ + kt + kh16 + 8,        av0, 8);
    ld8_h(Vp + (size_t)(16 + l15) * SEQ + kt + kh16,     av1, 0);
    ld8_h(Vp + (size_t)(16 + l15) * SEQ + kt + kh16 + 8, av1, 8);
    o0 = wmma_f16(av0, bp, o0);
    o1 = wmma_f16(av1, bp, o1);
  }
  float inv = 1.0f / lrun;
  const int q = (qtile << 4) + l15;
  _Float16* dst = ao + ((size_t)b * SEQ + q) * DIM + h * DH;
#pragma unroll
  for (int r = 0; r < 8; ++r) {
    dst[r + kh8]      = (_Float16)(o0[r] * inv);
    dst[16 + r + kh8] = (_Float16)(o1[r] * inv);
  }
}

// ---- phase 3: out-proj (WMMA) + bias + residual + LayerNorm ----------------
__global__ __launch_bounds__(256) void outln_kernel(
    const _Float16* __restrict__ ao, const float* __restrict__ Wf,
    const float* __restrict__ bfv, const float* __restrict__ X,
    const float* __restrict__ gamma, const float* __restrict__ beta,
    float* __restrict__ out) {
  __shared__ float xs[16 * 256];
  const int lane = threadIdx.x & 31;
  const int wid  = threadIdx.x >> 5;
  const int m0   = blockIdx.x << 4;
  const int l15  = lane & 15;
  const int hf   = lane >> 4;
  const int kh8  = hf << 3;
  const int kh16 = hf << 4;
  const _Float16* arow = ao + (size_t)(m0 + l15) * DIM;
#pragma unroll
  for (int t = 0; t < 2; ++t) {
    const int n0 = (wid * 2 + t) << 4;
    const int nB = n0 + l15;
    const float* wrow = Wf + (size_t)nB * DIM;
    v8f acc = {};
#pragma unroll
    for (int kt = 0; kt < DIM; kt += 32) {
      v16h a, bm;
      ld8_h(arow + kt + kh8,        a, 0);
      ld8_h(arow + kt + 16 + kh8,   a, 8);
      ld8_f2h(wrow + kt + kh8,      bm, 0);
      ld8_f2h(wrow + kt + 16 + kh8, bm, 8);
      acc = wmma_f16(a, bm, acc);
    }
    float bias = bfv[nB];
#pragma unroll
    for (int r = 0; r < 8; ++r) {
      int mloc = r + (hf << 3);
      xs[mloc * 256 + nB] = acc[r] + bias + X[(size_t)(m0 + mloc) * DIM + nB];
    }
  }
  __syncthreads();
  // LayerNorm: wave handles 2 rows, 16 lanes/row, butterfly within lane-halves
  const int rrow = wid * 2 + hf;
  const float* xr = xs + rrow * 256;
  float sum = 0.0f;
#pragma unroll
  for (int j = 0; j < 16; ++j) sum += xr[l15 + 16 * j];
  sum += __shfl_xor(sum, 1, 32);
  sum += __shfl_xor(sum, 2, 32);
  sum += __shfl_xor(sum, 4, 32);
  sum += __shfl_xor(sum, 8, 32);
  float mu = sum * (1.0f / 256.0f);
  float vs = 0.0f;
#pragma unroll
  for (int j = 0; j < 16; ++j) { float d = xr[l15 + 16 * j] - mu; vs += d * d; }
  vs += __shfl_xor(vs, 1, 32);
  vs += __shfl_xor(vs, 2, 32);
  vs += __shfl_xor(vs, 4, 32);
  vs += __shfl_xor(vs, 8, 32);
  float rstd = rsqrtf(vs * (1.0f / 256.0f) + 1e-6f);
  float* orow = out + (size_t)(m0 + rrow) * DIM;
#pragma unroll
  for (int j = 0; j < 16; ++j) {
    int c = l15 + 16 * j;
    orow[c] = (xr[c] - mu) * rstd * gamma[c] + beta[c];
  }
}

// ---- launch ----------------------------------------------------------------
extern "C" void kernel_launch(void* const* d_in, const int* in_sizes, int n_in,
                              void* d_out, int out_size, void* d_ws, size_t ws_size,
                              hipStream_t stream) {
  const float* X  = (const float*)d_in[0];
  const float* Wq = (const float*)d_in[1];
  const float* Wk = (const float*)d_in[2];
  const float* Wv = (const float*)d_in[3];
  const float* Wf = (const float*)d_in[4];
  const float* bf = (const float*)d_in[5];
  const float* g  = (const float*)d_in[6];
  const float* be = (const float*)d_in[7];
  float* out = (float*)d_out;

  const size_t n = (size_t)NBATCH * SEQ * DIM;   // 2M elems
  _Float16* qb  = (_Float16*)d_ws;               // 4 MiB
  _Float16* kb  = qb + n;                        // 4 MiB
  _Float16* vtb = kb + n;                        // 4 MiB (transposed V)
  _Float16* ao  = vtb + n;                       // 4 MiB (attention out, f16)

  qkv_kernel<<<dim3(3072), dim3(256), 0, stream>>>(X, Wq, Wk, Wv, qb, kb, vtb);
  attn_kernel<<<dim3(512), dim3(256), 0, stream>>>(qb, kb, vtb, ao);
  outln_kernel<<<dim3(512), dim3(256), 0, stream>>>(ao, Wf, bf, X, g, be, out);
}
